// TrackReconstructionModel_73315091743317
// MI455X (gfx1250) — matrix operand order repaired and checked by reading, not compiled, not run on hardware
//
#include <hip/hip_runtime.h>
#include <hip/hip_bf16.h>

#define DCH 256
#define NH 8
#define HD 32
#define DFF 512
#define NHITS 16384
#define NQR 2100
#define NLE 12
#define NLD 8
#define WH 128
#define CHUNK 128
#define LWIN 384
#define NKP 2112

typedef __attribute__((ext_vector_type(16))) __bf16 bf16x16;
typedef __attribute__((ext_vector_type(8)))  float  floatx8;
typedef __attribute__((ext_vector_type(4)))  unsigned uint4v;
typedef __attribute__((ext_vector_type(8)))  unsigned uint8v;

__device__ __forceinline__ floatx8 wmma_bf16(bf16x16 a, bf16x16 b, floatx8 c) {
  return __builtin_amdgcn_wmma_f32_16x16x32_bf16(false, a, false, b, (short)0, c, false, false);
}
__device__ __forceinline__ int a_kmap(int i, int half) { return i + half * 16; }
__device__ __forceinline__ floatx8 zero8() {
  floatx8 v;
#pragma unroll
  for (int i = 0; i < 8; ++i) v[i] = 0.f;
  return v;
}

// ---------------------------------------------------------------- small utils
__global__ void minmax_kernel(const float* __restrict__ coords, float* __restrict__ mm, int n) {
  __shared__ float s0[256], s1[256], s2[256], s3[256];
  float rmin = 3.4e38f, rmax = -3.4e38f, zmin = 3.4e38f, zmax = -3.4e38f;
  for (int i = threadIdx.x; i < n; i += 256) {
    float r = coords[i * 3 + 0], z = coords[i * 3 + 2];
    rmin = fminf(rmin, r); rmax = fmaxf(rmax, r);
    zmin = fminf(zmin, z); zmax = fmaxf(zmax, z);
  }
  s0[threadIdx.x] = rmin; s1[threadIdx.x] = rmax; s2[threadIdx.x] = zmin; s3[threadIdx.x] = zmax;
  __syncthreads();
  for (int o = 128; o > 0; o >>= 1) {
    if (threadIdx.x < (unsigned)o) {
      s0[threadIdx.x] = fminf(s0[threadIdx.x], s0[threadIdx.x + o]);
      s1[threadIdx.x] = fmaxf(s1[threadIdx.x], s1[threadIdx.x + o]);
      s2[threadIdx.x] = fminf(s2[threadIdx.x], s2[threadIdx.x + o]);
      s3[threadIdx.x] = fmaxf(s3[threadIdx.x], s3[threadIdx.x + o]);
    }
    __syncthreads();
  }
  if (threadIdx.x == 0) { mm[0] = s0[0]; mm[1] = s1[0]; mm[2] = s2[0]; mm[3] = s3[0]; }
}

__global__ void embed_kernel(const float* __restrict__ hits, const float* __restrict__ coords,
                             const float* __restrict__ emb_w, const float* __restrict__ emb_b,
                             const float* __restrict__ mm, float* __restrict__ x, int n) {
  int t = blockIdx.x * 256 + threadIdx.x;
  if (t >= n * DCH) return;
  int row = t / DCH, c = t % DCH;
  float acc = emb_b[c];
#pragma unroll
  for (int k = 0; k < 16; ++k) acc += hits[row * 16 + k] * emb_w[k * DCH + c];
  float phi = coords[row * 3 + 1];
  if (c == 0) acc += __sinf(phi);
  else if (c == 1) acc += __cosf(phi);
  else if (c == 2) acc += (coords[row * 3 + 0] - mm[0]) / (mm[1] - mm[0] + 1e-8f);
  else if (c == 3) acc += (coords[row * 3 + 2] - mm[2]) / (mm[3] - mm[2] + 1e-8f);
  x[t] = acc;
}

__global__ void sort_init(const float* __restrict__ coords, float* __restrict__ keys,
                          int* __restrict__ idx, int n) {
  int i = blockIdx.x * 256 + threadIdx.x;
  if (i < n) { keys[i] = coords[i * 3 + 1]; idx[i] = i; }
}
__global__ void bitonic_step(float* __restrict__ keys, int* __restrict__ idx, int j, int k, int n) {
  int i = blockIdx.x * 256 + threadIdx.x;
  if (i >= n) return;
  int l = i ^ j;
  if (l <= i) return;
  bool up = ((i & k) == 0);
  float ki = keys[i], kl = keys[l];
  if ((ki > kl) == up) {
    keys[i] = kl; keys[l] = ki;
    int t = idx[i]; idx[i] = idx[l]; idx[l] = t;
  }
}

__global__ void gather_rows(const float* __restrict__ src, const int* __restrict__ idx,
                            float* __restrict__ dst, int rows) {
  int t = blockIdx.x * 256 + threadIdx.x;
  if (t >= rows * DCH) return;
  int r = t / DCH, c = t % DCH;
  dst[t] = src[(size_t)idx[r] * DCH + c];
}
__global__ void scatter_rows(const float* __restrict__ src, const int* __restrict__ idx,
                             float* __restrict__ dst, int rows) {
  int t = blockIdx.x * 256 + threadIdx.x;
  if (t >= rows * DCH) return;
  int r = t / DCH, c = t % DCH;
  dst[(size_t)idx[r] * DCH + c] = src[t];
}
__global__ void copy_kernel(const float* __restrict__ src, float* __restrict__ dst, int n) {
  int t = blockIdx.x * 256 + threadIdx.x;
  if (t < n) dst[t] = src[t];
}

// LayerNorm: one row (256 cols) per wave32, 8 rows per 256-thread block.
__global__ void ln_kernel(const float* __restrict__ x, const float* __restrict__ g,
                          const float* __restrict__ b, float* __restrict__ out, int rows) {
  int wave = threadIdx.x >> 5, lane = threadIdx.x & 31;
  int row = blockIdx.x * 8 + wave;
  if (row >= rows) return;
  const float* xr = x + (size_t)row * DCH;
  float v[8], s = 0.f;
#pragma unroll
  for (int i = 0; i < 8; ++i) { v[i] = xr[lane + i * 32]; s += v[i]; }
#pragma unroll
  for (int o = 16; o >= 1; o >>= 1) s += __shfl_xor(s, o, 32);
  float mu = s * (1.f / DCH);
  float vs = 0.f;
#pragma unroll
  for (int i = 0; i < 8; ++i) { float d = v[i] - mu; vs += d * d; }
#pragma unroll
  for (int o = 16; o >= 1; o >>= 1) vs += __shfl_xor(vs, o, 32);
  float rs = rsqrtf(vs * (1.f / DCH) + 1e-5f);
#pragma unroll
  for (int i = 0; i < 8; ++i) {
    int c = lane + i * 32;
    out[(size_t)row * DCH + c] = (v[i] - mu) * rs * g[c] + b[c];
  }
}

// ---------------------------------------------------------------- generic WMMA GEMM
// C = epilogue(A[MxK] @ B[KxN or NxK if TB] + bias) [* mul] [+ res]
// block 128 (4 waves), 64x64 tile per block, k-step 32.
template <int ACT, bool TB>
__global__ void gemm_wmma_kernel(const float* __restrict__ A, const float* __restrict__ B,
                                 const float* __restrict__ bias, const float* __restrict__ mul,
                                 const float* __restrict__ res, float* __restrict__ C,
                                 int M, int N, int K) {
  __shared__ __bf16 sA[64 * 32];
  __shared__ __bf16 sB[32 * 64];
  const int tid = threadIdx.x;
  const int wave = tid >> 5, lane = tid & 31;
  const int half = lane >> 4, nl = lane & 15;
  const int m0 = blockIdx.y * 64, n0 = blockIdx.x * 64;
  floatx8 acc[4];
#pragma unroll
  for (int i = 0; i < 4; ++i) acc[i] = zero8();
  for (int k0 = 0; k0 < K; k0 += 32) {
    for (int e = tid; e < 64 * 32; e += 128) {
      int r = e >> 5, c = e & 31;
      int gr = m0 + r;
      sA[e] = (__bf16)((gr < M) ? A[(size_t)gr * K + k0 + c] : 0.f);
    }
    for (int e = tid; e < 32 * 64; e += 128) {
      int r = e >> 6, c = e & 63;
      int gc = n0 + c;
      float v = 0.f;
      if (gc < N) v = TB ? B[(size_t)gc * K + k0 + r] : B[(size_t)(k0 + r) * N + gc];
      sB[e] = (__bf16)v;
    }
    if (k0 + 32 < K) {  // hint next A tile toward the caches (global_prefetch_b8)
      int gr = m0 + (tid >> 1);
      if (gr < M) __builtin_prefetch(&A[(size_t)gr * K + k0 + 32], 0, 1);
    }
    __syncthreads();
    bf16x16 af;
#pragma unroll
    for (int i = 0; i < 16; ++i) af[i] = sA[(wave * 16 + nl) * 32 + a_kmap(i, half)];
#pragma unroll
    for (int nt = 0; nt < 4; ++nt) {
      bf16x16 bf;
#pragma unroll
      for (int i = 0; i < 16; ++i) bf[i] = sB[(half * 16 + i) * 64 + nt * 16 + nl];
      acc[nt] = wmma_bf16(af, bf, acc[nt]);
    }
    __syncthreads();
  }
#pragma unroll
  for (int nt = 0; nt < 4; ++nt) {
#pragma unroll
    for (int g = 0; g < 8; ++g) {
      int row = m0 + wave * 16 + g + half * 8;
      int col = n0 + nt * 16 + nl;
      if (row < M && col < N) {
        float v = acc[nt][g];
        if (bias) v += bias[col];
        if (ACT == 1) v = fmaxf(v, 0.f);
        else if (ACT == 2) v = v / (1.f + __expf(-v));
        else if (ACT == 3) v = 1.f / (1.f + __expf(-v));
        size_t o = (size_t)row * N + col;
        if (mul) v *= mul[o];
        if (res) v += res[o];
        C[o] = v;
      }
    }
  }
}

// ---------------------------------------------------------------- encoder sliding attention
// grid(nc=128, 2, H=8), block 128 (4 waves). 64 query rows per block, 384-key window.
__global__ void sliding_attn_kernel(const float* __restrict__ Qo, const float* __restrict__ Ko,
                                    const float* __restrict__ Vo, float* __restrict__ Os, int n) {
  extern __shared__ char smem[];
  __bf16* sQ = (__bf16*)smem;          // 64x32
  __bf16* sK = sQ + 64 * 32;           // 384x32
  __bf16* sV = sK + LWIN * 32;         // 384x32
  float* sS = (float*)(sV + LWIN * 32);  // 64x384
  const int j = blockIdx.x, hb = blockIdx.y, h = blockIdx.z;
  const int tid = threadIdx.x, wave = tid >> 5, lane = tid & 31;
  const int half = lane >> 4, nl = lane & 15;
  for (int e = tid; e < 64 * HD; e += 128) {
    int r = e >> 5, c = e & 31;
    int gq = j * CHUNK + hb * 64 + r;
    sQ[e] = (__bf16)Qo[(size_t)gq * DCH + h * HD + c];
  }
  for (int e = tid; e < LWIN * HD; e += 128) {
    int r = e >> 5, c = e & 31;
    int gk = (j * CHUNK + r - WH + n) % n;
    sK[e] = (__bf16)Ko[(size_t)gk * DCH + h * HD + c];
    sV[e] = (__bf16)Vo[(size_t)gk * DCH + h * HD + c];
  }
  __syncthreads();
  bf16x16 af;
#pragma unroll
  for (int i = 0; i < 16; ++i) af[i] = sQ[(wave * 16 + nl) * HD + a_kmap(i, half)];
  const float scale = 0.17677669529663687f;  // 1/sqrt(32)
  for (int nt = 0; nt < LWIN / 16; ++nt) {
    bf16x16 bf;
#pragma unroll
    for (int i = 0; i < 16; ++i) bf[i] = sK[(nt * 16 + nl) * HD + half * 16 + i];
    floatx8 sc = wmma_bf16(af, bf, zero8());
#pragma unroll
    for (int g = 0; g < 8; ++g) {
      int rl = wave * 16 + g + half * 8;
      int rc = hb * 64 + rl;   // row within chunk (0..127)
      int m = nt * 16 + nl;    // key position (0..383)
      bool valid = (unsigned)(m - rc) <= (unsigned)(2 * WH);  // m>=rc && m<=rc+256
      sS[rl * LWIN + m] = valid ? sc[g] * scale : -1e30f;
    }
  }
  __syncthreads();
  if (tid < 64) {
    float* row = sS + tid * LWIN;
    float mx = -1e30f;
    for (int m = 0; m < LWIN; ++m) mx = fmaxf(mx, row[m]);
    float sm = 0.f;
    for (int m = 0; m < LWIN; ++m) { float p = __expf(row[m] - mx); row[m] = p; sm += p; }
    float inv = 1.f / sm;
    for (int m = 0; m < LWIN; ++m) row[m] *= inv;
  }
  __syncthreads();
  floatx8 o0 = zero8(), o1 = zero8();
  for (int ks = 0; ks < LWIN / 32; ++ks) {
    bf16x16 pa;
#pragma unroll
    for (int i = 0; i < 16; ++i) pa[i] = (__bf16)sS[(wave * 16 + nl) * LWIN + ks * 32 + a_kmap(i, half)];
    bf16x16 b0, b1;
#pragma unroll
    for (int i = 0; i < 16; ++i) {
      int kr = ks * 32 + half * 16 + i;
      b0[i] = sV[kr * HD + nl];
      b1[i] = sV[kr * HD + 16 + nl];
    }
    o0 = wmma_bf16(pa, b0, o0);
    o1 = wmma_bf16(pa, b1, o1);
  }
#pragma unroll
  for (int g = 0; g < 8; ++g) {
    int rl = wave * 16 + g + half * 8;
    int gq = j * CHUNK + hb * 64 + rl;
    Os[(size_t)gq * DCH + h * HD + nl] = o0[g];
    Os[(size_t)gq * DCH + h * HD + 16 + nl] = o1[g];
  }
}

// ---------------------------------------------------------------- decoder self-attention
// grid(ceil(nq/16), H), block 64 (2 waves). Full 2112-wide score row in LDS.
__global__ void dec_self_attn_kernel(const float* __restrict__ Qd, const float* __restrict__ Kd,
                                     const float* __restrict__ Vd, float* __restrict__ O, int nq) {
  extern __shared__ char smem[];
  float* sS = (float*)smem;              // 16*2112
  __bf16* sQ = (__bf16*)(sS + 16 * NKP); // 16*32
  __bf16* sT = sQ + 16 * HD;             // 32*32 staging
  float* red = (float*)(sT + 32 * HD);
  float* pmax = red;         // 64
  float* rmax = red + 64;    // 16
  float* psum = red + 80;    // 64
  float* rsum = red + 144;   // 16
  const int qb = blockIdx.x, h = blockIdx.y;
  const int tid = threadIdx.x, wave = tid >> 5, lane = tid & 31;
  const int half = lane >> 4, nl = lane & 15;
  for (int e = tid; e < 16 * HD; e += 64) {
    int r = e >> 5, c = e & 31;
    int gq = qb * 16 + r;
    sQ[e] = (__bf16)((gq < nq) ? Qd[(size_t)gq * DCH + h * HD + c] : 0.f);
  }
  __syncthreads();
  bf16x16 af;
#pragma unroll
  for (int i = 0; i < 16; ++i) af[i] = sQ[nl * HD + a_kmap(i, half)];
  const float scale = 0.17677669529663687f;
  for (int kt = 0; kt < NKP / 32; ++kt) {
    for (int e = tid; e < 32 * HD; e += 64) {
      int r = e >> 5, c = e & 31;
      int gk = kt * 32 + r;
      sT[e] = (__bf16)((gk < nq) ? Kd[(size_t)gk * DCH + h * HD + c] : 0.f);
    }
    __syncthreads();
    bf16x16 bf;
#pragma unroll
    for (int i = 0; i < 16; ++i) bf[i] = sT[(wave * 16 + nl) * HD + half * 16 + i];
    floatx8 sc = wmma_bf16(af, bf, zero8());
#pragma unroll
    for (int g = 0; g < 8; ++g) {
      int rl = g + half * 8;
      int col = kt * 32 + wave * 16 + nl;
      sS[rl * NKP + col] = (col < nq) ? sc[g] * scale : -1e30f;
    }
    __syncthreads();
  }
  {  // softmax over 2112 cols (unnormalized; divide at the end)
    int r = tid >> 2, part = tid & 3;
    float mx = -1e30f;
    for (int c = part; c < NKP; c += 4) mx = fmaxf(mx, sS[r * NKP + c]);
    pmax[r * 4 + part] = mx;
    __syncthreads();
    if (tid < 16)
      rmax[tid] = fmaxf(fmaxf(pmax[tid * 4], pmax[tid * 4 + 1]),
                        fmaxf(pmax[tid * 4 + 2], pmax[tid * 4 + 3]));
    __syncthreads();
    float rm = rmax[r], sm = 0.f;
    for (int c = part; c < NKP; c += 4) {
      float p = __expf(sS[r * NKP + c] - rm);
      sS[r * NKP + c] = p;
      sm += p;
    }
    psum[r * 4 + part] = sm;
    __syncthreads();
    if (tid < 16)
      rsum[tid] = psum[tid * 4] + psum[tid * 4 + 1] + psum[tid * 4 + 2] + psum[tid * 4 + 3];
    __syncthreads();
  }
  floatx8 acc = zero8();
  for (int ks = 0; ks < NKP / 32; ++ks) {
    for (int e = tid; e < 32 * HD; e += 64) {
      int r = e >> 5, c = e & 31;
      int gk = ks * 32 + r;
      sT[e] = (__bf16)((gk < nq) ? Vd[(size_t)gk * DCH + h * HD + c] : 0.f);
    }
    __syncthreads();
    bf16x16 pa, bf;
#pragma unroll
    for (int i = 0; i < 16; ++i) pa[i] = (__bf16)sS[nl * NKP + ks * 32 + a_kmap(i, half)];
#pragma unroll
    for (int i = 0; i < 16; ++i) bf[i] = sT[(half * 16 + i) * HD + wave * 16 + nl];
    acc = wmma_bf16(pa, bf, acc);
    __syncthreads();
  }
#pragma unroll
  for (int g = 0; g < 8; ++g) {
    int rl = g + half * 8;
    int gq = qb * 16 + rl;
    if (gq < nq) O[(size_t)gq * DCH + h * HD + wave * 16 + nl] = acc[g] / rsum[rl];
  }
}

// ---------------------------------------------------------------- decoder cross-attention (flash)
// Fuses: s = qn@hitE^T/16 [+ logsigmoid(qmask@hitE^T)], online softmax, q += P@hitE.
// grid(ceil(nq/16)), block 256 (8 waves). Waves 0-3 score-WMMA, waves 4-7 mask-WMMA,
// every wave owns a 32-column slice of the output accumulator.
// hitE tiles (64x256 fp32) are staged into LDS by the Tensor Data Mover:
// wave 0 issues tensor_load_to_lds with a D# built per ISA ch.8, waits on
// TENSORcnt, then the workgroup barrier releases the consumer waves.
__global__ void cross_attn_kernel(float* __restrict__ q, const float* __restrict__ qn,
                                  const float* __restrict__ qmask, const float* __restrict__ hitE,
                                  int nq, int nk) {
  extern __shared__ char smem[];
  float* sHit = (float*)smem;                 // 64x256 fp32 (TDM target, offset 0)
  __bf16* sQn = (__bf16*)(sHit + 64 * DCH);   // 16x256
  __bf16* sQm = sQn + 16 * DCH;               // 16x256
  float* sS = (float*)(sQm + 16 * DCH);       // 16x64
  float* sM = sS + 16 * 64;                   // 16x64
  __bf16* sP = (__bf16*)(sM + 16 * 64);       // 16x64
  float* red = (float*)(sP + 16 * 64);
  float* pmax = red;          // 64
  float* psum = red + 64;     // 64
  float* rmax = red + 128;    // 16
  float* rsum = red + 144;    // 16
  float* rscale = red + 160;  // 16
  const int qb = blockIdx.x;
  const int tid = threadIdx.x, wave = tid >> 5, lane = tid & 31;
  const int half = lane >> 4, nl = lane & 15;
  const bool useMask = (qmask != nullptr);
  for (int e = tid; e < 16 * DCH; e += 256) {
    int r = e >> 8, c = e & 255;
    int gq = qb * 16 + r;
    sQn[e] = (__bf16)((gq < nq) ? qn[(size_t)gq * DCH + c] : 0.f);
    sQm[e] = (__bf16)((useMask && gq < nq) ? qmask[(size_t)gq * DCH + c] : 0.f);
  }
  if (tid < 16) { rmax[tid] = -1e30f; rsum[tid] = 0.f; }
  floatx8 acc0 = zero8(), acc1 = zero8();
  const float scale = 0.0625f;  // 1/sqrt(256)
  for (int t = 0; t < nk / 64; ++t) {
    __syncthreads();  // previous tile fully consumed before DMA overwrites it
    if (wave == 0) {
      unsigned long long ga = (unsigned long long)(const void*)(hitE + (size_t)t * 64 * DCH);
      uint4v g0;
      g0[0] = 1u;                                   // count=1 (valid user descriptor)
      g0[1] = 0u;                                   // lds_addr: sHit at base of dynamic LDS
      g0[2] = (unsigned)ga;                         // global_addr[31:0]
      g0[3] = (unsigned)((ga >> 32) & 0x1FFFFFFull) | (2u << 30);  // addr[56:32] | type=2
      uint8v g1;
      g1[0] = 2u << 16;                             // data_size = 4 bytes
      g1[1] = (unsigned)(DCH & 0xFFFF) << 16;       // tensor_dim0[15:0] at bits 63:48
      g1[2] = (unsigned)(DCH >> 16) | ((unsigned)(nk & 0xFFFF) << 16);  // dim0 hi | dim1 lo
      g1[3] = (unsigned)(nk >> 16) | ((unsigned)DCH << 16);            // dim1 hi | tile_dim0
      g1[4] = 64u;                                  // tile_dim1=64, tile_dim2=0
      g1[5] = (unsigned)DCH;                        // tensor_dim0_stride[31:0]
      g1[6] = 0u;                                   // stride0 hi | dim1_stride lo
      g1[7] = 0u;                                   // dim1_stride hi
      asm volatile("tensor_load_to_lds %0, %1" :: "s"(g0), "s"(g1) : "memory");
      __builtin_amdgcn_s_wait_tensorcnt(0);
    }
    __syncthreads();
    if (wave < 4) {
      floatx8 sc = zero8();
#pragma unroll
      for (int ks = 0; ks < 8; ++ks) {
        bf16x16 a, b;
#pragma unroll
        for (int i = 0; i < 16; ++i) a[i] = sQn[nl * DCH + ks * 32 + a_kmap(i, half)];
#pragma unroll
        for (int i = 0; i < 16; ++i)
          b[i] = (__bf16)sHit[(wave * 16 + nl) * DCH + ks * 32 + half * 16 + i];
        sc = wmma_bf16(a, b, sc);
      }
#pragma unroll
      for (int g = 0; g < 8; ++g) sS[(g + half * 8) * 64 + wave * 16 + nl] = sc[g];
    } else if (useMask) {
      int w = wave - 4;
      floatx8 sc = zero8();
#pragma unroll
      for (int ks = 0; ks < 8; ++ks) {
        bf16x16 a, b;
#pragma unroll
        for (int i = 0; i < 16; ++i) a[i] = sQm[nl * DCH + ks * 32 + a_kmap(i, half)];
#pragma unroll
        for (int i = 0; i < 16; ++i)
          b[i] = (__bf16)sHit[(w * 16 + nl) * DCH + ks * 32 + half * 16 + i];
        sc = wmma_bf16(a, b, sc);
      }
#pragma unroll
      for (int g = 0; g < 8; ++g) sM[(g + half * 8) * 64 + w * 16 + nl] = sc[g];
    }
    __syncthreads();
    if (tid < 64) {  // combine + local max
      int r = tid >> 2, part = tid & 3;
      float mx = -1e30f;
      for (int c = part * 16; c < part * 16 + 16; ++c) {
        float sv = sS[r * 64 + c] * scale;
        if (useMask) {
          float z = sM[r * 64 + c];  // log(sigmoid(z)) = -softplus(-z)
          sv += (z > 0.f) ? -log1pf(__expf(-z)) : (z - log1pf(__expf(z)));
        }
        sS[r * 64 + c] = sv;
        mx = fmaxf(mx, sv);
      }
      pmax[r * 4 + part] = mx;
    }
    __syncthreads();
    if (tid < 16) {
      float m4 = fmaxf(fmaxf(pmax[tid * 4], pmax[tid * 4 + 1]),
                       fmaxf(pmax[tid * 4 + 2], pmax[tid * 4 + 3]));
      float nm = fmaxf(rmax[tid], m4);
      rscale[tid] = __expf(rmax[tid] - nm);
      rmax[tid] = nm;
    }
    __syncthreads();
    if (tid < 64) {
      int r = tid >> 2, part = tid & 3;
      float rm = rmax[r], sm = 0.f;
      for (int c = part * 16; c < part * 16 + 16; ++c) {
        float p = __expf(sS[r * 64 + c] - rm);
        sP[r * 64 + c] = (__bf16)p;
        sm += p;
      }
      psum[r * 4 + part] = sm;
    }
    __syncthreads();
    if (tid < 16)
      rsum[tid] = rsum[tid] * rscale[tid] +
                  psum[tid * 4] + psum[tid * 4 + 1] + psum[tid * 4 + 2] + psum[tid * 4 + 3];
    __syncthreads();
#pragma unroll
    for (int g = 0; g < 8; ++g) {
      float f = rscale[g + half * 8];
      acc0[g] *= f; acc1[g] *= f;
    }
#pragma unroll
    for (int ks = 0; ks < 2; ++ks) {
      bf16x16 pa;
#pragma unroll
      for (int i = 0; i < 16; ++i) pa[i] = sP[nl * 64 + ks * 32 + a_kmap(i, half)];
      bf16x16 b0, b1;
#pragma unroll
      for (int i = 0; i < 16; ++i) {
        int kr = ks * 32 + half * 16 + i;
        b0[i] = (__bf16)sHit[kr * DCH + wave * 32 + nl];
        b1[i] = (__bf16)sHit[kr * DCH + wave * 32 + 16 + nl];
      }
      acc0 = wmma_bf16(pa, b0, acc0);
      acc1 = wmma_bf16(pa, b1, acc1);
    }
  }
  __syncthreads();
#pragma unroll
  for (int g = 0; g < 8; ++g) {
    int rl = g + half * 8;
    int gq = qb * 16 + rl;
    if (gq < nq) {
      float inv = 1.f / rsum[rl];
      q[(size_t)gq * DCH + wave * 32 + nl] += acc0[g] * inv;
      q[(size_t)gq * DCH + wave * 32 + 16 + nl] += acc1[g] * inv;
    }
  }
}

// ---------------------------------------------------------------- host orchestration
static void gemm(hipStream_t s, int act, bool tb, const float* A, const float* B,
                 const float* bias, const float* mul, const float* res, float* C,
                 int M, int N, int K) {
  dim3 g((N + 63) / 64, (M + 63) / 64), b(128);
#define GCASE(a, t) gemm_wmma_kernel<a, t><<<g, b, 0, s>>>(A, B, bias, mul, res, C, M, N, K)
  if (!tb) {
    switch (act) {
      case 0: GCASE(0, false); break;
      case 1: GCASE(1, false); break;
      case 2: GCASE(2, false); break;
      default: GCASE(3, false); break;
    }
  } else {
    switch (act) {
      case 0: GCASE(0, true); break;
      case 1: GCASE(1, true); break;
      case 2: GCASE(2, true); break;
      default: GCASE(3, true); break;
    }
  }
#undef GCASE
}

extern "C" void kernel_launch(void* const* d_in, const int* in_sizes, int n_in,
                              void* d_out, int out_size, void* d_ws, size_t ws_size,
                              hipStream_t stream) {
  (void)in_sizes; (void)n_in; (void)out_size; (void)ws_size;
  const int N = NHITS, NQ2 = NQR;
#define IN(i) ((const float*)d_in[i])
  const float* hits = IN(0);
  const float* coords = IN(1);
  const float* emb_w = IN(2);
  const float* emb_b = IN(3);
  // encoder params [4..21]
  const float *e_ln1g = IN(4), *e_ln1b = IN(5), *e_qw = IN(6), *e_qb = IN(7);
  const float *e_kw = IN(8), *e_kb = IN(9), *e_vw = IN(10), *e_vb = IN(11);
  const float *e_ow = IN(12), *e_ob = IN(13), *e_ln2g = IN(14), *e_ln2b = IN(15);
  const float *e_w1 = IN(16), *e_b1 = IN(17), *e_w2 = IN(18), *e_b2 = IN(19);
  const float *e_w3 = IN(20), *e_b3 = IN(21);
  const float* queries = IN(22);
  // decoder params [23..42]
  const float *d_ln1g = IN(23), *d_ln1b = IN(24), *d_sqw = IN(25), *d_sqb = IN(26);
  const float *d_skw = IN(27), *d_skb = IN(28), *d_svw = IN(29), *d_svb = IN(30);
  const float *d_sow = IN(31), *d_sob = IN(32), *d_ln2g = IN(33), *d_ln2b = IN(34);
  const float *d_ln3g = IN(35), *d_ln3b = IN(36), *d_w1 = IN(37), *d_b1 = IN(38);
  const float *d_w2 = IN(39), *d_b2 = IN(40), *d_w3 = IN(41), *d_b3 = IN(42);
  const float *hp_w = IN(43), *hp_b = IN(44);
  const float *cls_w1 = IN(45), *cls_b1 = IN(46), *cls_w2 = IN(47), *cls_b2 = IN(48);
  const float *mk_w1 = IN(49), *mk_b1 = IN(50), *mk_w2 = IN(51), *mk_b2 = IN(52);
  const float *pr_w1 = IN(53), *pr_b1 = IN(54), *pr_w2 = IN(55), *pr_b2 = IN(56);
  const float *pr_w3 = IN(57), *pr_b3 = IN(58);
#undef IN
  float* out = (float*)d_out;

  // workspace layout
  float* W = (float*)d_ws;
  size_t off = 0;
  auto alloc = [&](size_t n) { float* p = W + off; off += n; return p; };
  float* x = alloc((size_t)N * DCH);
  float* xn = alloc((size_t)N * DCH);
  float* xo = alloc((size_t)N * DCH);
  float* Qo = alloc((size_t)N * DCH);
  float* Ko = alloc((size_t)N * DCH);
  float* Vo = alloc((size_t)N * DCH);
  float* attnS = alloc((size_t)N * DCH);
  float* attnU = alloc((size_t)N * DCH);
  float* h1 = alloc((size_t)N * DFF);
  float* h2 = alloc((size_t)N * DFF);
  float* hitE = alloc((size_t)N * DCH);
  float* keys = alloc(N);
  int* idx = (int*)alloc(N);
  float* mm = alloc(4);
  float* q = alloc((size_t)NQ2 * DCH);
  float* qn = alloc((size_t)NQ2 * DCH);
  float* qmask = alloc((size_t)NQ2 * DCH);
  float* Qd = alloc((size_t)NQ2 * DCH);
  float* Kd = alloc((size_t)NQ2 * DCH);
  float* Vd = alloc((size_t)NQ2 * DCH);
  float* attq = alloc((size_t)NQ2 * DCH);
  float* t1 = alloc((size_t)NQ2 * DFF);
  float* t2 = alloc((size_t)NQ2 * DFF);
  float* memb = alloc((size_t)NQ2 * DCH);

  const int TPB = 256;
  const int gND = (N * DCH + TPB - 1) / TPB;
  const int gN = (N + TPB - 1) / TPB;

  // 1) embedding + positional encoding
  minmax_kernel<<<1, 256, 0, stream>>>(coords, mm, N);
  embed_kernel<<<gND, TPB, 0, stream>>>(hits, coords, emb_w, emb_b, mm, x, N);

  // 2) argsort(phi) via bitonic sort
  sort_init<<<gN, TPB, 0, stream>>>(coords, keys, idx, N);
  for (int k = 2; k <= N; k <<= 1)
    for (int j = k >> 1; j > 0; j >>= 1)
      bitonic_step<<<gN, TPB, 0, stream>>>(keys, idx, j, k, N);

  const size_t slidingLDS = (size_t)(64 * 32 + 2 * LWIN * 32) * sizeof(__bf16) +
                            (size_t)64 * LWIN * sizeof(float);
  // 3) encoder layers
  for (int l = 0; l < NLE; ++l) {
    ln_kernel<<<(N + 7) / 8, 256, 0, stream>>>(x, e_ln1g + l * DCH, e_ln1b + l * DCH, xn, N);
    gather_rows<<<gND, TPB, 0, stream>>>(xn, idx, xo, N);
    gemm(stream, 0, false, xo, e_qw + (size_t)l * DCH * DCH, e_qb + l * DCH, nullptr, nullptr, Qo, N, DCH, DCH);
    gemm(stream, 0, false, xo, e_kw + (size_t)l * DCH * DCH, e_kb + l * DCH, nullptr, nullptr, Ko, N, DCH, DCH);
    gemm(stream, 0, false, xo, e_vw + (size_t)l * DCH * DCH, e_vb + l * DCH, nullptr, nullptr, Vo, N, DCH, DCH);
    sliding_attn_kernel<<<dim3(N / CHUNK, 2, NH), 128, slidingLDS, stream>>>(Qo, Ko, Vo, attnS, N);
    scatter_rows<<<gND, TPB, 0, stream>>>(attnS, idx, attnU, N);
    gemm(stream, 0, false, attnU, e_ow + (size_t)l * DCH * DCH, e_ob + l * DCH, nullptr, x, x, N, DCH, DCH);
    ln_kernel<<<(N + 7) / 8, 256, 0, stream>>>(x, e_ln2g + l * DCH, e_ln2b + l * DCH, xn, N);
    gemm(stream, 2, false, xn, e_w1 + (size_t)l * DCH * DFF, e_b1 + l * DFF, nullptr, nullptr, h1, N, DFF, DCH);
    gemm(stream, 0, false, xn, e_w2 + (size_t)l * DCH * DFF, e_b2 + l * DFF, h1, nullptr, h2, N, DFF, DCH);
    gemm(stream, 0, false, h2, e_w3 + (size_t)l * DFF * DCH, e_b3 + l * DCH, nullptr, x, x, N, DCH, DFF);
  }

  // 4) hit embedding for decoder
  gemm(stream, 0, false, x, hp_w, hp_b, nullptr, nullptr, hitE, N, DCH, DCH);

  // 5) decoder
  copy_kernel<<<(NQ2 * DCH + TPB - 1) / TPB, TPB, 0, stream>>>(queries, q, NQ2 * DCH);
  const int QB = (NQ2 + 15) / 16;
  const size_t selfLDS = (size_t)16 * NKP * sizeof(float) +
                         (size_t)(16 * HD + 32 * HD) * sizeof(__bf16) + 160 * sizeof(float);
  const size_t crossLDS = (size_t)64 * DCH * sizeof(float) +
                          (size_t)(2 * 16 * DCH + 16 * 64) * sizeof(__bf16) +
                          (size_t)(2 * 16 * 64 + 176) * sizeof(float);
  for (int l = 0; l < NLD; ++l) {
    ln_kernel<<<(NQ2 + 7) / 8, 256, 0, stream>>>(q, d_ln1g + l * DCH, d_ln1b + l * DCH, qn, NQ2);
    gemm(stream, 0, false, qn, d_sqw + (size_t)l * DCH * DCH, d_sqb + l * DCH, nullptr, nullptr, Qd, NQ2, DCH, DCH);
    gemm(stream, 0, false, qn, d_skw + (size_t)l * DCH * DCH, d_skb + l * DCH, nullptr, nullptr, Kd, NQ2, DCH, DCH);
    gemm(stream, 0, false, qn, d_svw + (size_t)l * DCH * DCH, d_svb + l * DCH, nullptr, nullptr, Vd, NQ2, DCH, DCH);
    dec_self_attn_kernel<<<dim3(QB, NH), 64, selfLDS, stream>>>(Qd, Kd, Vd, attq, NQ2);
    gemm(stream, 0, false, attq, d_sow + (size_t)l * DCH * DCH, d_sob + l * DCH, nullptr, q, q, NQ2, DCH, DCH);
    ln_kernel<<<(NQ2 + 7) / 8, 256, 0, stream>>>(q, d_ln2g + l * DCH, d_ln2b + l * DCH, qn, NQ2);
    cross_attn_kernel<<<QB, 256, crossLDS, stream>>>(q, qn, (l == 0) ? nullptr : qmask, hitE, NQ2, N);
    ln_kernel<<<(NQ2 + 7) / 8, 256, 0, stream>>>(q, d_ln3g + l * DCH, d_ln3b + l * DCH, qn, NQ2);
    gemm(stream, 2, false, qn, d_w1 + (size_t)l * DCH * DFF, d_b1 + l * DFF, nullptr, nullptr, t1, NQ2, DFF, DCH);
    gemm(stream, 0, false, qn, d_w2 + (size_t)l * DCH * DFF, d_b2 + l * DFF, t1, nullptr, t2, NQ2, DFF, DCH);
    gemm(stream, 0, false, t2, d_w3 + (size_t)l * DFF * DCH, d_b3 + l * DCH, nullptr, q, q, NQ2, DCH, DFF);
    copy_kernel<<<(NQ2 * DCH + TPB - 1) / TPB, TPB, 0, stream>>>(q, qmask, NQ2 * DCH);
  }

  // 6) heads
  gemm(stream, 1, false, q, cls_w1, cls_b1, nullptr, nullptr, qn, NQ2, DCH, DCH);
  gemm(stream, 0, false, qn, cls_w2, cls_b2, nullptr, nullptr, out, NQ2, 1, DCH);
  gemm(stream, 1, false, q, mk_w1, mk_b1, nullptr, nullptr, t1, NQ2, DFF, DCH);
  gemm(stream, 0, false, t1, mk_w2, mk_b2, nullptr, nullptr, memb, NQ2, DCH, DFF);
  gemm(stream, 3, true, memb, hitE, nullptr, nullptr, nullptr, out + NQ2, NQ2, N, DCH);
  gemm(stream, 1, false, q, pr_w1, pr_b1, nullptr, nullptr, qn, NQ2, DCH, DCH);
  gemm(stream, 1, false, qn, pr_w2, pr_b2, nullptr, nullptr, t1, NQ2, DCH / 2, DCH);
  gemm(stream, 0, false, t1, pr_w3, pr_b3, nullptr, nullptr, out + NQ2 + (size_t)NQ2 * N, NQ2, 6, DCH / 2);
}
